// Attention_36163624632681
// MI455X (gfx1250) — matrix operand order repaired and checked by reading, not compiled, not run on hardware
//
#include <hip/hip_runtime.h>
#include <hip/hip_bf16.h>

// B=4, L1=L2=2048, D=1024, H=16, Dh=64, SCALE=1/8
#define B_   4
#define L1_  2048
#define L2_  2048
#define D_   1024
#define H_   16
#define DH_  64
#define SCALE_ 0.125f

#if defined(__has_builtin)
#  if __has_builtin(__builtin_amdgcn_global_load_async_to_lds_b128)
#    define USE_ASYNC_LDS 1
#  endif
#endif
#ifndef USE_ASYNC_LDS
#  define USE_ASYNC_LDS 0
#endif

typedef __attribute__((ext_vector_type(16))) __bf16 v16bf;
typedef __attribute__((ext_vector_type(8)))  float  v8f;

// Exact parameter type of the async-LDS builtins (from hipcc diagnostic):
//   "__attribute__((__vector_size__(4 * sizeof(int)))) int __device__ *"
typedef int async_b128_t __attribute__((vector_size(16)));
#define GPTR_B128(p) ((__attribute__((address_space(1))) async_b128_t*)(void*)(p))
#define LPTR_B128(p) ((__attribute__((address_space(3))) async_b128_t*)(void*)(p))

__device__ __forceinline__ v8f wmma_bf16(v16bf a, v16bf b, v8f c) {
  // D = A(16x32 bf16) * B(32x16 bf16) + C(16x16 f32)
  return __builtin_amdgcn_wmma_f32_16x16x32_bf16(false, a, false, b, (short)0, c, false, false);
}

// ISA 7.12.2: 16-bit A-matrix 16x32. lane -> M = lane%16; half h -> K:
//   K = (h>=8 ? 16:0) + (lane>=16 ? 8:0) + ((h>>1)&3)*2 + (h&1)
// Per lane: two contiguous 8-half (16B) runs -> vectorizes to ds_load_b128.
__device__ __forceinline__ int kmapA(int h, int lane) {
  return ((h >> 3) << 4) + ((lane & 16) ? 8 : 0) + (((h >> 1) & 3) << 1) + (h & 1);
}
// 16-bit B-matrix 32x16. lane -> N = lane%16; half h -> K = (h>=8 ? 16:0) + (lane>=16 ? 8:0) + (h&7)
// Per lane: two contiguous 8-half (16B) runs when LDS tile is stored N-major.
__device__ __forceinline__ int kmapB(int h, int lane) {
  return ((h >> 3) << 4) + ((lane & 16) ? 8 : 0) + (h & 7);
}
// f32 C/D 16x16: reg r -> M = r + (lane>=16 ? 8:0); N = lane%16

__device__ __forceinline__ void async_wait0() {
#if USE_ASYNC_LDS
#  if __has_builtin(__builtin_amdgcn_s_wait_asynccnt)
  __builtin_amdgcn_s_wait_asynccnt(0);
#  else
  asm volatile("s_wait_asynccnt 0x0" ::: "memory");
#  endif
#endif
}

// ---------------------------------------------------------------------------
// Projection GEMM: out[m][n] = bf16( sum_k X[m][k] * W[n][k] + bias[n] )
// Block = 128 threads (4 waves), 64x64 tile; K-step = 32.
// Both LDS tiles stored so that all WMMA fragment reads are contiguous.
// ---------------------------------------------------------------------------
__global__ __launch_bounds__(128)
void proj_gemm(const float* __restrict__ X, const float* __restrict__ W,
               const float* __restrict__ bias, __bf16* __restrict__ out) {
  __shared__ __align__(16) __bf16 Xs[64][40];  // [m][k]  (A is M-per-lane)
  __shared__ __align__(16) __bf16 Wn[64][40];  // [n][k]  (B is N-per-lane)

  const int t    = threadIdx.x;
  const int lane = t & 31;
  const int wave = t >> 5;
  const int n0   = blockIdx.x * 64;
  const int m0   = blockIdx.y * 64;

  v8f acc[4] = {};

  for (int kt = 0; kt < D_; kt += 32) {
    __syncthreads();
    // Stage X tile 64x32 (fp32 -> bf16): 512 float4 / 128 threads.
#pragma unroll
    for (int i = 0; i < 4; ++i) {
      int idx = t * 4 + i;
      int row = idx >> 3;
      int c4  = (idx & 7) << 2;
      float4 xv = *(const float4*)(X + (size_t)(m0 + row) * D_ + kt + c4);
      Xs[row][c4 + 0] = (__bf16)xv.x;
      Xs[row][c4 + 1] = (__bf16)xv.y;
      Xs[row][c4 + 2] = (__bf16)xv.z;
      Xs[row][c4 + 3] = (__bf16)xv.w;
    }
    // Stage W tile rows n0..n0+63 x cols kt..kt+31, kept N-major (no transpose).
#pragma unroll
    for (int i = 0; i < 4; ++i) {
      int idx = t * 4 + i;
      int n   = idx >> 3;
      int c4  = (idx & 7) << 2;
      float4 wv = *(const float4*)(W + (size_t)(n0 + n) * D_ + kt + c4);
      Wn[n][c4 + 0] = (__bf16)wv.x;
      Wn[n][c4 + 1] = (__bf16)wv.y;
      Wn[n][c4 + 2] = (__bf16)wv.z;
      Wn[n][c4 + 3] = (__bf16)wv.w;
    }
    __syncthreads();

    v16bf a;
    int mrow = wave * 16 + (lane & 15);
#pragma unroll
    for (int h = 0; h < 16; ++h) a[h] = Xs[mrow][kmapA(h, lane)];

#pragma unroll
    for (int nt = 0; nt < 4; ++nt) {
      v16bf bb;
      int ncol = nt * 16 + (lane & 15);
#pragma unroll
      for (int h = 0; h < 16; ++h) bb[h] = Wn[ncol][kmapB(h, lane)];
      acc[nt] = wmma_bf16(a, bb, acc[nt]);
    }
  }

#pragma unroll
  for (int nt = 0; nt < 4; ++nt) {
    int n = n0 + nt * 16 + (lane & 15);
    float bv = bias[n];
#pragma unroll
    for (int r = 0; r < 8; ++r) {
      int m = m0 + wave * 16 + r + ((lane & 16) ? 8 : 0);
      out[(size_t)m * D_ + n] = (__bf16)(acc[nt][r] + bv);
    }
  }
}

// ---------------------------------------------------------------------------
// Flash attention per (b, h, 64-query tile). 128 threads = 4 waves.
// K tile staged row-major (async LDS copy when available); V tile staged
// TRANSPOSED ([dim][key]) so P*V B-fragments are contiguous ds_load_b128.
// Online softmax tracks m, l1=sum(exp), l2=sum(exp^2):
//   out = O / l1 (rescale==0);  out = O / sqrt(l2) (rescale==1)
// ---------------------------------------------------------------------------
__global__ __launch_bounds__(128)
void flash_attn(const __bf16* __restrict__ Q, const __bf16* __restrict__ K,
                const __bf16* __restrict__ V, float* __restrict__ out,
                const int* __restrict__ rescale_flag) {
  __shared__ __align__(16) __bf16 Ks[64][72];      // [key][dim]
  __shared__ __align__(16) __bf16 Vt[64][72];      // [dim][key]  (transposed)
  __shared__ __align__(16) __bf16 Ps[4][16][72];   // per-wave P staging [m][key]

  const int t    = threadIdx.x;
  const int lane = t & 31;
  const int wave = t >> 5;
  const int qt   = blockIdx.x;
  const int h    = blockIdx.y;
  const int b    = blockIdx.z;
  const int rescale = *rescale_flag;

  // Q A-fragments, resident for the whole kernel (dims 0-31 / 32-63).
  v16bf qa0, qa1;
  {
    int l = qt * 64 + wave * 16 + (lane & 15);
    const __bf16* qrow = Q + ((size_t)(b * L1_ + l)) * D_ + (size_t)h * DH_;
#pragma unroll
    for (int hh = 0; hh < 16; ++hh) {
      qa0[hh] = qrow[kmapA(hh, lane)];
      qa1[hh] = qrow[32 + kmapA(hh, lane)];
    }
  }

  v8f o[4] = {};
  float mrow[8], l1s[8], l2s[8];
#pragma unroll
  for (int r = 0; r < 8; ++r) { mrow[r] = -3.0e38f; l1s[r] = 0.f; l2s[r] = 0.f; }

  for (int l20 = 0; l20 < L2_; l20 += 64) {
    __syncthreads();

    // ---- Stage K tile [64 keys][64 dims], row-major (async if available).
    {
#pragma unroll
      for (int i = 0; i < 4; ++i) {
        int c   = t * 4 + i;          // 0..511 16B-chunks
        int row = c >> 3;
        int off = (c & 7) << 3;       // element offset (x8 bf16 per chunk)
        const __bf16* gp =
            K + ((size_t)(b * L2_ + l20 + row)) * D_ + (size_t)h * DH_ + off;
#if USE_ASYNC_LDS
        __builtin_amdgcn_global_load_async_to_lds_b128(
            GPTR_B128(const_cast<__bf16*>(gp)),
            LPTR_B128(&Ks[row][off]), 0, 0);
#else
        *(uint4*)&Ks[row][off] = *(const uint4*)gp;
#endif
      }
    }
    // ---- Stage V tile transposed: thread owns 4 keys x 8 dims.
    {
      int q0 = (t >> 3) << 2;   // key base (0..60)
      int d0 = (t & 7) << 3;    // dim base (0..56)
      uint4 raw[4];
#pragma unroll
      for (int i = 0; i < 4; ++i)
        raw[i] = *(const uint4*)(V + ((size_t)(b * L2_ + l20 + q0 + i)) * D_ +
                                 (size_t)h * DH_ + d0);
      const unsigned short* rs = (const unsigned short*)raw;
#pragma unroll
      for (int j = 0; j < 8; ++j) {
        ushort4 pk = make_ushort4(rs[0 * 8 + j], rs[1 * 8 + j],
                                  rs[2 * 8 + j], rs[3 * 8 + j]);
        *(ushort4*)(void*)&Vt[d0 + j][q0] = pk;
      }
    }
    // Prefetch next K/V tiles into cache while this tile computes.
    if (l20 + 64 < L2_) {
      size_t nb = ((size_t)(b * L2_ + l20 + 64 + (t >> 1))) * D_ +
                  (size_t)h * DH_ + (t & 1) * 32;
      __builtin_prefetch(K + nb, 0, 1);
      __builtin_prefetch(V + nb, 0, 1);
    }
    async_wait0();
    __syncthreads();

    // ---- S = Q * K^T over 4 key sub-tiles (2 chained WMMAs each; dh=64).
    v8f s[4];
#pragma unroll
    for (int nt = 0; nt < 4; ++nt) {
      v16bf b0, b1;
      int nrow = nt * 16 + (lane & 15);
#pragma unroll
      for (int hh = 0; hh < 16; ++hh) {
        b0[hh] = Ks[nrow][kmapB(hh, lane)];
        b1[hh] = Ks[nrow][32 + kmapB(hh, lane)];
      }
      v8f z = {};
      z = wmma_bf16(qa0, b0, z);
      z = wmma_bf16(qa1, b1, z);
      s[nt] = z;
    }

    // ---- Online softmax (row r spans a 16-lane group).
#pragma unroll
    for (int r = 0; r < 8; ++r) {
      float mx = -3.0e38f;
#pragma unroll
      for (int nt = 0; nt < 4; ++nt) {
        float vv = s[nt][r] * SCALE_;
        s[nt][r] = vv;
        mx = fmaxf(mx, vv);
      }
#pragma unroll
      for (int msk = 1; msk < 16; msk <<= 1) mx = fmaxf(mx, __shfl_xor(mx, msk, 32));
      float mnew  = fmaxf(mrow[r], mx);
      float alpha = __expf(mrow[r] - mnew);
      mrow[r] = mnew;
      l1s[r] *= alpha;
      l2s[r] *= alpha * alpha;
#pragma unroll
      for (int nt = 0; nt < 4; ++nt) o[nt][r] *= alpha;
      float ps1 = 0.f, ps2 = 0.f;
#pragma unroll
      for (int nt = 0; nt < 4; ++nt) {
        float p = __expf(s[nt][r] - mnew);
        s[nt][r] = p;
        ps1 += p;
        ps2 += p * p;
      }
#pragma unroll
      for (int msk = 1; msk < 16; msk <<= 1) {
        ps1 += __shfl_xor(ps1, msk, 32);
        ps2 += __shfl_xor(ps2, msk, 32);
      }
      l1s[r] += ps1;
      l2s[r] += ps2;
    }

    // ---- Re-layout P: C-fragment -> per-wave LDS tile -> A-fragment.
#pragma unroll
    for (int nt = 0; nt < 4; ++nt)
#pragma unroll
      for (int r = 0; r < 8; ++r)
        Ps[wave][r + ((lane & 16) ? 8 : 0)][nt * 16 + (lane & 15)] = (__bf16)s[nt][r];
    __syncthreads();

    v16bf pa0, pa1;
#pragma unroll
    for (int hh = 0; hh < 16; ++hh) {
      pa0[hh] = Ps[wave][lane & 15][kmapA(hh, lane)];
      pa1[hh] = Ps[wave][lane & 15][32 + kmapA(hh, lane)];
    }
    // ---- O += P * V : B(k=key, n=dim) = Vt[dim][key] -> contiguous reads.
#pragma unroll
    for (int nt = 0; nt < 4; ++nt) {
      v16bf b0, b1;
      int ncol = nt * 16 + (lane & 15);
#pragma unroll
      for (int hh = 0; hh < 16; ++hh) {
        b0[hh] = Vt[ncol][kmapB(hh, lane)];
        b1[hh] = Vt[ncol][32 + kmapB(hh, lane)];
      }
      o[nt] = wmma_bf16(pa0, b0, o[nt]);
      o[nt] = wmma_bf16(pa1, b1, o[nt]);
    }
  }

  // ---- Epilogue: out = O / (rescale ? sqrt(l2) : l1)
#pragma unroll
  for (int r = 0; r < 8; ++r) {
    float denom = rescale ? sqrtf(l2s[r]) : l1s[r];
    float inv   = 1.0f / denom;
    int l = qt * 64 + wave * 16 + r + ((lane & 16) ? 8 : 0);
    float* orow = out + ((size_t)(b * L1_ + l)) * D_ + h * DH_;
#pragma unroll
    for (int nt = 0; nt < 4; ++nt)
      orow[nt * 16 + (lane & 15)] = o[nt][r] * inv;
  }
}

// ---------------------------------------------------------------------------
extern "C" void kernel_launch(void* const* d_in, const int* in_sizes, int n_in,
                              void* d_out, int out_size, void* d_ws, size_t ws_size,
                              hipStream_t stream) {
  const float* inp1 = (const float*)d_in[0];
  const float* inp2 = (const float*)d_in[1];
  const float* Wq   = (const float*)d_in[2];
  const float* bq   = (const float*)d_in[3];
  const float* Wk   = (const float*)d_in[4];
  const float* bk   = (const float*)d_in[5];
  const float* Wv   = (const float*)d_in[6];
  const float* bv   = (const float*)d_in[7];
  const int*   rsc  = (const int*)d_in[8];
  float* out = (float*)d_out;

  // Workspace: Q/K/V projections in bf16 (3 x 8192x1024 x 2B = 48 MB).
  __bf16* qws = (__bf16*)d_ws;
  __bf16* kws = qws + (size_t)B_ * L1_ * D_;
  __bf16* vws = kws + (size_t)B_ * L2_ * D_;

  dim3 pgrid(D_ / 64, (B_ * L1_) / 64);
  proj_gemm<<<pgrid, 128, 0, stream>>>(inp1, Wq, bq, qws);
  proj_gemm<<<pgrid, 128, 0, stream>>>(inp2, Wk, bk, kws);
  proj_gemm<<<pgrid, 128, 0, stream>>>(inp2, Wv, bv, vws);

  dim3 agrid(L1_ / 64, H_, B_);
  flash_attn<<<agrid, 128, 0, stream>>>(qws, kws, vws, out, rsc);
}
